// EnhancedGraphSAGE_10050223473234
// MI455X (gfx1250) — compile-verified, matrix operand order checked by reading, not run on hardware
//
#include <hip/hip_runtime.h>
#include <hip/hip_bf16.h>

typedef __attribute__((ext_vector_type(16))) _Float16 v16h;
typedef __attribute__((ext_vector_type(8)))  float    v8f;

#define N_NODES 100000
#define C_IN    128
#define C_HID   128
#define C_OUT   64
#define N_EDGES 1600000
#define N_PAIR  200000

__device__ __forceinline__ v8f wmma16(v16h a, v16h b, v8f c) {
  // D = A(16x32 f16) * B(32x16 f16) + C(16x16 f32)
  return __builtin_amdgcn_wmma_f32_16x16x32_f16(false, a, false, b, (short)0, c, false, false);
}

// Build A fragment for one K-step (32 k's). Lane layout (CDNA5 ISA):
// lane l: M = l&15, hl = l>>4; half j -> k = kb + j (j<8) / kb + 16 + (j-8),
// where kb = kstep*32 + hl*8. Both 8-runs are contiguous in the source row.
__device__ __forceinline__ v16h load_a_frag(const float* __restrict__ rowp, int kb, float scale) {
  const float4 p0 = *(const float4*)(rowp + kb);
  const float4 p1 = *(const float4*)(rowp + kb + 4);
  const float4 q0 = *(const float4*)(rowp + kb + 16);
  const float4 q1 = *(const float4*)(rowp + kb + 20);
  v16h a;
  a[0]  = (_Float16)(p0.x * scale); a[1]  = (_Float16)(p0.y * scale);
  a[2]  = (_Float16)(p0.z * scale); a[3]  = (_Float16)(p0.w * scale);
  a[4]  = (_Float16)(p1.x * scale); a[5]  = (_Float16)(p1.y * scale);
  a[6]  = (_Float16)(p1.z * scale); a[7]  = (_Float16)(p1.w * scale);
  a[8]  = (_Float16)(q0.x * scale); a[9]  = (_Float16)(q0.y * scale);
  a[10] = (_Float16)(q0.z * scale); a[11] = (_Float16)(q0.w * scale);
  a[12] = (_Float16)(q1.x * scale); a[13] = (_Float16)(q1.y * scale);
  a[14] = (_Float16)(q1.z * scale); a[15] = (_Float16)(q1.w * scale);
  return a;
}

// B fragments are pre-packed so each lane reads 16 contiguous halves (32B).
__device__ __forceinline__ v16h load_b_frag(const _Float16* __restrict__ packed, int fragid, int lane) {
  return *(const v16h*)(packed + ((size_t)fragid << 9) + (lane << 4));
}

// ---------------- weight pre-pack: W[k][n] f32 -> fragment-order f16 -------
// fragid = ks*ntiles + nt ; within fragment: lane (0..31), half j (0..15)
// n = nt*16 + (lane&15); k = ks*32 + (lane>>4)*8 + j + (j>=8 ? 8 : 0)
__global__ void pack_weights(const float* __restrict__ Wa, const float* __restrict__ Wb,
                             _Float16* __restrict__ dst, int K1, int Ktot, int Cout) {
  int i = blockIdx.x * blockDim.x + threadIdx.x;
  int total = Ktot * Cout;
  if (i >= total) return;
  int fragid = i >> 9;
  int r      = i & 511;
  int lane   = r >> 4;
  int j      = r & 15;
  int ntiles = Cout >> 4;
  int ks = fragid / ntiles;
  int nt = fragid - ks * ntiles;
  int n  = nt * 16 + (lane & 15);
  int k  = ks * 32 + (lane >> 4) * 8 + j + ((j >= 8) ? 8 : 0);
  float v = (k < K1) ? Wa[k * Cout + n] : Wb[(k - K1) * Cout + n];
  dst[i] = (_Float16)v;
}

// ---------------- misc ------------------------------------------------------
__global__ void zero_f(float* __restrict__ p, size_t n) {
  size_t i = (size_t)blockIdx.x * blockDim.x + threadIdx.x;
  size_t st = (size_t)gridDim.x * blockDim.x;
  for (; i < n; i += st) p[i] = 0.0f;
}

__global__ void scatter_deg(const int* __restrict__ ei, float* __restrict__ deg, int E) {
  int e = blockIdx.x * blockDim.x + threadIdx.x;
  if (e < E) unsafeAtomicAdd(&deg[ei[E + e]], 1.0f);
}

// one wave per edge; each lane moves 4 channels (32*4 = 128)
__global__ void scatter_feat(const int* __restrict__ ei, const float* __restrict__ x,
                             float* __restrict__ agg, int E) {
  int gid = blockIdx.x * blockDim.x + threadIdx.x;
  int e = gid >> 5;
  if (e >= E) return;
  int c = (gid & 31) << 2;
  int s = ei[e];
  int d = ei[E + e];
  const float4 v = *(const float4*)(x + (size_t)s * C_IN + c);
  float* ap = agg + (size_t)d * C_IN + c;
  unsafeAtomicAdd(ap + 0, v.x);
  unsafeAtomicAdd(ap + 1, v.y);
  unsafeAtomicAdd(ap + 2, v.z);
  unsafeAtomicAdd(ap + 3, v.w);
}

// ---------------- fused SAGE layer GEMM ------------------------------------
// out = [mean | x] @ [Wl;Wr](packed) + bl  ; optional BN(eval)+ReLU+residual
// one wave per 16-row tile; K = 256 (8 K-steps); all Cout column tiles
__global__ __launch_bounds__(32)
void sage_gemm(const float* __restrict__ agg, const float* __restrict__ deg,
               const float* __restrict__ xin, const _Float16* __restrict__ Bp,
               const float* __restrict__ bias, const float* __restrict__ gamma,
               const float* __restrict__ beta, float* __restrict__ out,
               int Cout, int flags /* bit0: BN+ReLU, bit1: residual */) {
  const int lane = threadIdx.x & 31;
  const int hl   = lane >> 4;
  const int m    = lane & 15;
  const int row  = blockIdx.x * 16 + m;
  const float rdeg = 1.0f / fmaxf(deg[row], 1.0f);
  const float* aggRow = agg + (size_t)row * C_IN;
  const float* xRow   = xin + (size_t)row * C_IN;

  v16h afr[8];
#pragma unroll
  for (int ks = 0; ks < 8; ++ks) {
    int kb = ks * 32 + hl * 8;
    afr[ks] = (kb < C_IN) ? load_a_frag(aggRow, kb, rdeg)
                          : load_a_frag(xRow, kb - C_IN, 1.0f);
  }

  const int ntiles = Cout >> 4;
  for (int nt = 0; nt < ntiles; ++nt) {
    v8f acc = {0.f, 0.f, 0.f, 0.f, 0.f, 0.f, 0.f, 0.f};
#pragma unroll
    for (int ks = 0; ks < 8; ++ks)
      acc = wmma16(afr[ks], load_b_frag(Bp, ks * ntiles + nt, lane), acc);

    const int col = nt * 16 + m;
    const float bb = bias[col];
    float sc = 1.0f, sh = 0.0f;
    if (flags & 1) { sc = gamma[col] * rsqrtf(1.0f + 1e-5f); sh = beta[col]; }
#pragma unroll
    for (int r = 0; r < 8; ++r) {
      const int rowm = blockIdx.x * 16 + hl * 8 + r;   // C/D layout
      float v = acc[r] + bb;
      if (flags & 1) v = fmaxf(v * sc + sh, 0.0f);
      if (flags & 2) v += xin[(size_t)rowm * C_IN + col];
      out[(size_t)rowm * Cout + col] = v;
    }
  }
}

// ---------------- fused edge decoder ---------------------------------------
// per wave: 16 edges. hidden = relu([z[s]|z[d]] @ We1 + be1); out = sigmoid(hidden.We2 + be2)
__global__ __launch_bounds__(32)
void decode_edges(const float* __restrict__ z, const int* __restrict__ ei,
                  const _Float16* __restrict__ Bp, const float* __restrict__ be1,
                  const float* __restrict__ We2, const float* __restrict__ be2,
                  float* __restrict__ out, int P) {
  const int lane = threadIdx.x & 31;
  const int hl   = lane >> 4;
  const int m    = lane & 15;
  const int e    = blockIdx.x * 16 + m;
  const int s = ei[e];
  const int d = ei[P + e];
  const float* zs = z + (size_t)s * C_OUT;
  const float* zd = z + (size_t)d * C_OUT;

  v16h afr[4];
#pragma unroll
  for (int ks = 0; ks < 4; ++ks) {
    int kb = ks * 32 + hl * 8;
    afr[ks] = (kb < C_OUT) ? load_a_frag(zs, kb, 1.0f)
                           : load_a_frag(zd, kb - C_OUT, 1.0f);
  }

  float partial[8] = {0.f, 0.f, 0.f, 0.f, 0.f, 0.f, 0.f, 0.f};
#pragma unroll
  for (int nt = 0; nt < 8; ++nt) {
    v8f acc = {0.f, 0.f, 0.f, 0.f, 0.f, 0.f, 0.f, 0.f};
#pragma unroll
    for (int ks = 0; ks < 4; ++ks)
      acc = wmma16(afr[ks], load_b_frag(Bp, ks * 8 + nt, lane), acc);
    const int col = nt * 16 + m;
    const float bb = be1[col];
    const float w2 = We2[col];
#pragma unroll
    for (int r = 0; r < 8; ++r)
      partial[r] += fmaxf(acc[r] + bb, 0.0f) * w2;
  }

  const float b2 = be2[0];
#pragma unroll
  for (int r = 0; r < 8; ++r) {
    float v = partial[r];
    v += __shfl_xor(v, 1, 16);
    v += __shfl_xor(v, 2, 16);
    v += __shfl_xor(v, 4, 16);
    v += __shfl_xor(v, 8, 16);
    if (m == 0) {
      const int ee = blockIdx.x * 16 + hl * 8 + r;     // rows 0-7 in lane0, 8-15 in lane16
      out[ee] = 1.0f / (1.0f + __expf(-(v + b2)));
    }
  }
}

// ---------------- host orchestration ---------------------------------------
extern "C" void kernel_launch(void* const* d_in, const int* in_sizes, int n_in,
                              void* d_out, int out_size, void* d_ws, size_t ws_size,
                              hipStream_t stream) {
  const float* x   = (const float*)d_in[0];
  const int* ei    = (const int*)d_in[1];
  const int* pei   = (const int*)d_in[2];
  const int* nei   = (const int*)d_in[3];
  const float* Wl1 = (const float*)d_in[4];
  const float* bl1 = (const float*)d_in[5];
  const float* Wr1 = (const float*)d_in[6];
  const float* g1  = (const float*)d_in[7];
  const float* b1  = (const float*)d_in[8];
  const float* Wl2 = (const float*)d_in[9];
  const float* bl2 = (const float*)d_in[10];
  const float* Wr2 = (const float*)d_in[11];
  const float* g2  = (const float*)d_in[12];
  const float* b2  = (const float*)d_in[13];
  const float* Wl3 = (const float*)d_in[14];
  const float* bl3 = (const float*)d_in[15];
  const float* Wr3 = (const float*)d_in[16];
  const float* We1 = (const float*)d_in[17];
  const float* be1 = (const float*)d_in[18];
  const float* We2 = (const float*)d_in[19];
  const float* be2 = (const float*)d_in[20];

  float* z   = (float*)d_out;              // [100000, 64]
  float* pos = z + (size_t)N_NODES * C_OUT;
  float* neg = pos + N_PAIR;

  // workspace layout (all offsets 32B-aligned)
  float* deg = (float*)d_ws;                                  // N
  float* agg = deg + N_NODES;                                 // N*128
  float* h1  = agg + (size_t)N_NODES * C_IN;                  // N*128
  float* h2  = h1  + (size_t)N_NODES * C_IN;                  // N*128
  _Float16* W1p  = (_Float16*)(h2 + (size_t)N_NODES * C_IN);  // 256*128
  _Float16* W2p  = W1p + 256 * 128;                           // 256*128
  _Float16* W3p  = W2p + 256 * 128;                           // 256*64
  _Float16* We1p = W3p + 256 * 64;                            // 128*128

  // pack weights into fragment-order f16
  pack_weights<<<(256 * 128 + 255) / 256, 256, 0, stream>>>(Wl1, Wr1, W1p, 128, 256, 128);
  pack_weights<<<(256 * 128 + 255) / 256, 256, 0, stream>>>(Wl2, Wr2, W2p, 128, 256, 128);
  pack_weights<<<(256 * 64  + 255) / 256, 256, 0, stream>>>(Wl3, Wr3, W3p, 128, 256, 64);
  pack_weights<<<(128 * 128 + 255) / 256, 256, 0, stream>>>(We1, We1, We1p, 128, 128, 128);

  // degree (same for all layers)
  zero_f<<<1024, 256, 0, stream>>>(deg, (size_t)N_NODES);
  scatter_deg<<<(N_EDGES + 255) / 256, 256, 0, stream>>>(ei, deg, N_EDGES);

  const size_t aggN = (size_t)N_NODES * C_IN;
  const int sgrid = (N_EDGES * 32) / 256;
  // layer 1: in = x, out = h1 (BN+ReLU+residual)
  zero_f<<<2048, 256, 0, stream>>>(agg, aggN);
  scatter_feat<<<sgrid, 256, 0, stream>>>(ei, x, agg, N_EDGES);
  sage_gemm<<<N_NODES / 16, 32, 0, stream>>>(agg, deg, x, W1p, bl1, g1, b1, h1, 128, 3);
  // layer 2: in = h1, out = h2
  zero_f<<<2048, 256, 0, stream>>>(agg, aggN);
  scatter_feat<<<sgrid, 256, 0, stream>>>(ei, h1, agg, N_EDGES);
  sage_gemm<<<N_NODES / 16, 32, 0, stream>>>(agg, deg, h1, W2p, bl2, g2, b2, h2, 128, 3);
  // layer 3: in = h2, out = z (no BN/ReLU/residual)
  zero_f<<<2048, 256, 0, stream>>>(agg, aggN);
  scatter_feat<<<sgrid, 256, 0, stream>>>(ei, h2, agg, N_EDGES);
  sage_gemm<<<N_NODES / 16, 32, 0, stream>>>(agg, deg, h2, W3p, bl3, bl3, bl3, z, 64, 0);

  // decoder
  decode_edges<<<N_PAIR / 16, 32, 0, stream>>>(z, pei, We1p, be1, We2, be2, pos, N_PAIR);
  decode_edges<<<N_PAIR / 16, 32, 0, stream>>>(z, nei, We1p, be1, We2, be2, neg, N_PAIR);
}
